// GroupedQueryAttention_72911364817031
// MI455X (gfx1250) — compile-verified, matrix operand order checked by reading, not run on hardware
//
#include <hip/hip_runtime.h>
#include <hip/hip_bf16.h>

// GQA forward for MI455X (gfx1250, wave32, WMMA f32_16x16x32_f16)
#define B_   2
#define S_   2048
#define D_   1024
#define H_   16
#define KVH_ 4
#define DH_  64
#define G_   4
#define KV_  256
#define BS_  (B_ * S_)

typedef __attribute__((ext_vector_type(16))) _Float16 v16h;
typedef __attribute__((ext_vector_type(8)))  _Float16 v8h;
typedef __attribute__((ext_vector_type(8)))  float    v8f;
typedef __attribute__((ext_vector_type(4)))  float    v4f;

// ---- fragment loaders -------------------------------------------------------
// A (16x32 f16) layout: lanes 0-15 hold row M=lane, K {0..7,16..23};
// lanes 16-31 hold row M=lane-16, K {8..15,24..31}.
// Dense B (32x16) layout == A-layout of B^T, so the same loader serves both
// when B is stored transposed (row-major B^T).
__device__ inline v16h frag_from_f16(const _Float16* p, int half) {
  v8h lo = *(const v8h*)(p + half * 8);
  v8h hi = *(const v8h*)(p + 16 + half * 8);
  v16h r;
#pragma unroll
  for (int i = 0; i < 8; ++i) { r[i] = lo[i]; r[8 + i] = hi[i]; }
  return r;
}
__device__ inline v16h frag_from_f32(const float* p, int half) {
  v4f a0 = *(const v4f*)(p + half * 8);
  v4f a1 = *(const v4f*)(p + half * 8 + 4);
  v4f b0 = *(const v4f*)(p + 16 + half * 8);
  v4f b1 = *(const v4f*)(p + 16 + half * 8 + 4);
  v16h r;
#pragma unroll
  for (int i = 0; i < 4; ++i) {
    r[i]      = (_Float16)a0[i];
    r[4 + i]  = (_Float16)a1[i];
    r[8 + i]  = (_Float16)b0[i];
    r[12 + i] = (_Float16)b1[i];
  }
  return r;
}
__device__ inline v16h frag_load(const float* p, int half)    { return frag_from_f32(p, half); }
__device__ inline v16h frag_load(const _Float16* p, int half) { return frag_from_f16(p, half); }

__device__ inline v8f wmma_f16(v16h a, v16h b, v8f c) {
  // (neg_a, A, neg_b, B, c_mod, C, reuse_a, reuse_b)
  return __builtin_amdgcn_wmma_f32_16x16x32_f16(false, a, false, b, (short)0, c,
                                                false, false);
}

// ---- weight transpose + f32 -> f16 ------------------------------------------
__global__ void transpose_cvt(const float* __restrict__ W, _Float16* __restrict__ WT,
                              int kk, int nn) {
  int idx = blockIdx.x * blockDim.x + threadIdx.x;
  if (idx >= kk * nn) return;
  int kq = idx / nn;
  int n  = idx - kq * nn;
  WT[(size_t)n * kk + kq] = (_Float16)W[idx];
}

// ---- register-blocked WMMA GEMM: out = A[MxK] * (BT[NxK])^T -----------------
// 64x64 output tile per wave: 4 A-frags x 4 B-frags -> 16 WMMAs per k-chunk,
// 4x operand reuse (~21 FLOP/byte from L2 vs ~5 for a 16x16 tile).
// OMODE 0: f32 row-major [MxN]; 1: f16 row-major [MxN]; 2: f16 transposed [NxM]
template <typename AT, int OMODE>
__global__ __launch_bounds__(256) void gemm64(const AT* __restrict__ A,
                                              const _Float16* __restrict__ BT,
                                              void* __restrict__ out,
                                              int M, int N, int K) {
  int gw   = (int)((blockIdx.x * blockDim.x + threadIdx.x) >> 5);
  int lane = threadIdx.x & 31;
  int lid  = lane & 15;
  int half = lane >> 4;
  int tilesN = N >> 6;
  int tm = gw / tilesN;
  int tn = gw - tm * tilesN;
  if (tm * 64 >= M) return;  // wave-uniform

  const AT*       abase = A  + (size_t)(tm * 64 + lid) * K;
  const _Float16* bbase = BT + (size_t)(tn * 64 + lid) * K;
  const size_t    step16 = (size_t)16 * K;  // next 16-row group

  v8f acc[4][4];
  v8f zero = {};
#pragma unroll
  for (int mi = 0; mi < 4; ++mi)
#pragma unroll
    for (int ni = 0; ni < 4; ++ni) acc[mi][ni] = zero;

  for (int kc = 0; kc < K; kc += 32) {
    v16h af[4], bf[4];
#pragma unroll
    for (int mi = 0; mi < 4; ++mi)
      af[mi] = frag_load(abase + (size_t)mi * step16 + kc, half);
#pragma unroll
    for (int ni = 0; ni < 4; ++ni)
      bf[ni] = frag_from_f16(bbase + (size_t)ni * step16 + kc, half);
#pragma unroll
    for (int mi = 0; mi < 4; ++mi)
#pragma unroll
      for (int ni = 0; ni < 4; ++ni)
        acc[mi][ni] = wmma_f16(af[mi], bf[ni], acc[mi][ni]);
  }

  // C/D layout: VGPR r, lanes 0-15 -> (M=r, N=lane); lanes 16-31 -> (M=r+8, N=lane-16)
#pragma unroll
  for (int mi = 0; mi < 4; ++mi)
#pragma unroll
    for (int ni = 0; ni < 4; ++ni) {
      int gn = tn * 64 + ni * 16 + lid;
#pragma unroll
      for (int r = 0; r < 8; ++r) {
        int gm = tm * 64 + mi * 16 + r + 8 * half;
        if constexpr (OMODE == 0)
          ((float*)out)[(size_t)gm * N + gn] = acc[mi][ni][r];
        else if constexpr (OMODE == 1)
          ((_Float16*)out)[(size_t)gm * N + gn] = (_Float16)acc[mi][ni][r];
        else
          ((_Float16*)out)[(size_t)gn * M + gm] = (_Float16)acc[mi][ni][r];
      }
    }
}

// ---- flash-style causal attention: one wave per (b, head, 16-query tile) ----
__global__ __launch_bounds__(128) void attn_kernel(const _Float16* __restrict__ q,
                                                   const _Float16* __restrict__ k,
                                                   const _Float16* __restrict__ vT,
                                                   _Float16* __restrict__ ctx) {
  __shared__ _Float16 smem[4][16][32];  // per-wave P staging (C-layout -> A-layout)

  const int w    = threadIdx.x >> 5;
  const int lane = threadIdx.x & 31;
  const int lid  = lane & 15;
  const int half = lane >> 4;
  const int QT   = S_ / 16;

  int gw  = blockIdx.x * 4 + w;
  int qt  = gw % QT;
  int h   = (gw / QT) % H_;
  int b   = gw / (QT * H_);
  int kvh = h / G_;  // interleaved jnp.repeat -> head h reads kv head h/G

  // Q fragments (row-major f16, head slice), loaded once
  const _Float16* qrow = q + ((size_t)(b * S_ + qt * 16 + lid) * D_ + h * DH_);
  v16h aq0 = frag_from_f16(qrow, half);       // dims 0..31
  v16h aq1 = frag_from_f16(qrow + 32, half);  // dims 32..63

  v8f  o[4];
  v8f  zero = {};
#pragma unroll
  for (int c = 0; c < 4; ++c) o[c] = zero;
  float m_i[8], l_i[8];
#pragma unroll
  for (int r = 0; r < 8; ++r) { m_i[r] = -3.0e38f; l_i[r] = 0.f; }

  const int nkb = (qt * 16 + 47) / 32;  // causal: 32-key blocks up to the diagonal
  for (int jb = 0; jb < nkb; ++jb) {
    int k0 = jb * 32;

    // prefetch next key tile while we do softmax VALU work (global_prefetch path)
    if (jb + 1 < nkb) {
      const _Float16* pk =
          k + ((size_t)(b * S_ + k0 + 32 + lid) * KV_ + kvh * DH_);
      __builtin_prefetch(pk, 0, 3);
    }

    // scores = Q K^T / sqrt(64): two 16-key groups, 2 WMMAs each (K = 64 = 2x32)
    v8f s0, s1;
    {
      const _Float16* krow = k + ((size_t)(b * S_ + k0 + lid) * KV_ + kvh * DH_);
      v8f c = zero;
      c = wmma_f16(aq0, frag_from_f16(krow, half), c);
      c = wmma_f16(aq1, frag_from_f16(krow + 32, half), c);
      s0 = c;
    }
    {
      const _Float16* krow = k + ((size_t)(b * S_ + k0 + 16 + lid) * KV_ + kvh * DH_);
      v8f c = zero;
      c = wmma_f16(aq0, frag_from_f16(krow, half), c);
      c = wmma_f16(aq1, frag_from_f16(krow + 32, half), c);
      s1 = c;
    }

    // online softmax over the 32 columns (row = r + 8*half, col = lid / 16+lid)
    int key0 = k0 + lid, key1 = k0 + 16 + lid;
    float e0[8], e1[8], corr[8];
#pragma unroll
    for (int r = 0; r < 8; ++r) {
      int   row = qt * 16 + r + 8 * half;
      float v0  = (key0 <= row) ? s0[r] * 0.125f : -1.0e30f;
      float v1  = (key1 <= row) ? s1[r] * 0.125f : -1.0e30f;
      float mx  = fmaxf(v0, v1);
#pragma unroll
      for (int msk = 1; msk < 16; msk <<= 1) mx = fmaxf(mx, __shfl_xor(mx, msk, 32));
      float nm = fmaxf(m_i[r], mx);
      corr[r]  = __expf(m_i[r] - nm);
      float p0 = __expf(v0 - nm);
      float p1 = __expf(v1 - nm);
      float sm = p0 + p1;
#pragma unroll
      for (int msk = 1; msk < 16; msk <<= 1) sm += __shfl_xor(sm, msk, 32);
      l_i[r] = l_i[r] * corr[r] + sm;
      m_i[r] = nm;
      e0[r] = p0; e1[r] = p1;
    }

    // rescale running output
#pragma unroll
    for (int c = 0; c < 4; ++c)
#pragma unroll
      for (int r = 0; r < 8; ++r) o[c][r] *= corr[r];

    // stage P (f16) through LDS: C-layout write, A-layout read
#pragma unroll
    for (int r = 0; r < 8; ++r) {
      int rl = r + 8 * half;
      smem[w][rl][lid]      = (_Float16)e0[r];
      smem[w][rl][16 + lid] = (_Float16)e1[r];
    }
    asm volatile("s_wait_dscnt 0" ::: "memory");  // same-wave LDS RAW fence
    v16h pa = frag_from_f16(&smem[w][lid][0], half);

    // O += P * V : B^T rows come straight from the pre-transposed V^T [KV x BS]
#pragma unroll
    for (int c = 0; c < 4; ++c) {
      const _Float16* vrow =
          vT + ((size_t)(kvh * DH_ + c * 16 + lid) * BS_ + b * S_ + k0);
      o[c] = wmma_f16(pa, frag_from_f16(vrow, half), o[c]);
    }
  }

  // normalize and emit context (f16, [BS x D] row-major)
#pragma unroll
  for (int c = 0; c < 4; ++c)
#pragma unroll
    for (int r = 0; r < 8; ++r) {
      size_t row = (size_t)(b * S_ + qt * 16 + r + 8 * half);
      ctx[row * D_ + h * DH_ + c * 16 + lid] = (_Float16)(o[c][r] / l_i[r]);
    }
}

// ---- host-side orchestration ------------------------------------------------
extern "C" void kernel_launch(void* const* d_in, const int* in_sizes, int n_in,
                              void* d_out, int out_size, void* d_ws, size_t ws_size,
                              hipStream_t stream) {
  const float* hs = (const float*)d_in[0];
  const float* Wq = (const float*)d_in[1];
  const float* Wk = (const float*)d_in[2];
  const float* Wv = (const float*)d_in[3];
  const float* Wo = (const float*)d_in[4];

  _Float16* ws  = (_Float16*)d_ws;
  _Float16* wqT = ws;                          // [D  x D ]
  _Float16* wkT = wqT + (size_t)D_ * D_;       // [KV x D ]
  _Float16* wvT = wkT + (size_t)KV_ * D_;      // [KV x D ]
  _Float16* woT = wvT + (size_t)KV_ * D_;      // [D  x D ]
  _Float16* qb  = woT + (size_t)D_ * D_;       // [BS x D ]
  _Float16* kb  = qb  + (size_t)BS_ * D_;      // [BS x KV]
  _Float16* vT  = kb  + (size_t)BS_ * KV_;     // [KV x BS] (V transposed)
  _Float16* ctx = vT  + (size_t)KV_ * BS_;     // [BS x D ]

  // 1) weights -> f16, transposed (so B fragments are contiguous row loads)
  transpose_cvt<<<(D_ * D_) / 256, 256, 0, stream>>>(Wq, wqT, D_, D_);
  transpose_cvt<<<(D_ * KV_) / 256, 256, 0, stream>>>(Wk, wkT, D_, KV_);
  transpose_cvt<<<(D_ * KV_) / 256, 256, 0, stream>>>(Wv, wvT, D_, KV_);
  transpose_cvt<<<(D_ * D_) / 256, 256, 0, stream>>>(Wo, woT, D_, D_);

  // 2) projections (f32 A converted on load, f32 WMMA accumulation)
  {
    int tiles = (BS_ / 64) * (D_ / 64);
    gemm64<float, 1><<<tiles / 8, 256, 0, stream>>>(hs, wqT, qb, BS_, D_, D_);
  }
  {
    int tiles = (BS_ / 64) * (KV_ / 64);
    gemm64<float, 1><<<tiles / 8, 256, 0, stream>>>(hs, wkT, kb, BS_, KV_, D_);
    gemm64<float, 2><<<tiles / 8, 256, 0, stream>>>(hs, wvT, vT, BS_, KV_, D_);
  }

  // 3) causal GQA attention, flash softmax, all-WMMA
  attn_kernel<<<(B_ * H_ * (S_ / 16)) / 4, 128, 0, stream>>>(qb, kb, vT, ctx);

  // 4) output projection -> f32 d_out
  {
    int tiles = (BS_ / 64) * (D_ / 64);
    gemm64<_Float16, 0><<<tiles / 8, 256, 0, stream>>>(ctx, woT, d_out, BS_, D_, D_);
  }
}
